// DifferentialMultiHeadAttention_22771916603938
// MI455X (gfx1250) — matrix operand order repaired and checked by reading, not compiled, not run on hardware
//
#include <hip/hip_runtime.h>
#include <hip/hip_bf16.h>

typedef unsigned short ushort_t;
typedef unsigned int   uint_t;

typedef __attribute__((ext_vector_type(16))) __bf16 v16bf;
typedef __attribute__((ext_vector_type(8)))  float  v8f;

#define DIM   1024
#define HEADS 16
#define DH    64
#define SEQ   2048
#define BATCH 2
#define MTOT  (BATCH * SEQ)   // 4096 rows of the token matrix
#define SCALE 0.125f          // 1/sqrt(64)

// ---------------------------------------------------------------- helpers ----

__device__ __forceinline__ ushort_t f2bf(float f) {
  uint_t u = __float_as_uint(f);
  u += 0x7FFFu + ((u >> 16) & 1u);          // round-to-nearest-even
  return (ushort_t)(u >> 16);
}
__device__ __forceinline__ float bf2f(ushort_t h) {
  return __uint_as_float(((uint_t)h) << 16);
}

struct Frag {                 // 16 bf16 = 8 dwords, WMMA operand
  union { uint_t u[8]; v16bf v; };
};

// A-matrix fragment (16x32 bf16, ISA 7.12.2): lane L holds row m0+(L&15);
// VGPR v holds K = (v>>2)*16 + (L>>4)*8 + (v&3)*2 and K+1 (contiguous pair).
__device__ __forceinline__ Frag load_afrag(const ushort_t* __restrict__ base,
                                           int pitch, int m0, int k0) {
  Frag f;
  const int lane = threadIdx.x & 31;
  const int half = lane >> 4;
  const ushort_t* row = base + (size_t)(m0 + (lane & 15)) * pitch + k0;
#pragma unroll
  for (int v = 0; v < 8; ++v) {
    int k = (v >> 2) * 16 + half * 8 + (v & 3) * 2;
    f.u[v] = *(const uint_t*)(row + k);
  }
  return f;
}

// B-matrix fragment (32x16 bf16) read from a K-major ("transposed") buffer:
// element (k, n) lives at base[n*pitch + k].  lane L holds column n0+(L&15);
// VGPR v holds K = (v>>2)*16 + (L>>4)*8 + (v&3)*2 and K+1 — contiguous dword in this layout.
__device__ __forceinline__ Frag load_bfragT(const ushort_t* __restrict__ base,
                                            int pitch, int n0, int k0) {
  Frag f;
  const int lane = threadIdx.x & 31;
  const ushort_t* row = base + (size_t)(n0 + (lane & 15)) * pitch
                             + k0 + (lane >> 4) * 8;
#pragma unroll
  for (int v = 0; v < 8; ++v) f.u[v] = *(const uint_t*)(row + (v >> 2) * 16 + (v & 3) * 2);
  return f;
}

__device__ __forceinline__ v8f wmma_bf16(const Frag& a, const Frag& b, v8f c) {
  return __builtin_amdgcn_wmma_f32_16x16x32_bf16(
      false, a.v, false, b.v, (short)0, c, false, false);
}

// ------------------------------------------------------------ conversions ----

__global__ __launch_bounds__(256) void cvt_f32_bf16(const float* __restrict__ in,
                                                    ushort_t* __restrict__ out,
                                                    int n) {
  int i = blockIdx.x * 256 + threadIdx.x;
  if (i < n) out[i] = f2bf(in[i]);
}

// 1024x1024 fp32 -> bf16 with transpose via 32x32 LDS tile (coalesced both ways)
__global__ __launch_bounds__(256) void tconv1024(const float* __restrict__ in,
                                                 ushort_t* __restrict__ out) {
  __shared__ ushort_t tile[32][33];               // pad to kill bank conflicts
  const int tx = threadIdx.x & 31;                // fast-moving index
  const int ty = threadIdx.x >> 5;                // 0..7
  const int r0 = blockIdx.y * 32;
  const int c0 = blockIdx.x * 32;
#pragma unroll
  for (int i = 0; i < 4; ++i) {
    const int r = ty + i * 8;
    tile[tx][r] = f2bf(in[(size_t)(r0 + r) * DIM + c0 + tx]);   // coalesced read
  }
  __syncthreads();
#pragma unroll
  for (int i = 0; i < 4; ++i) {
    const int c = ty + i * 8;
    out[(size_t)(c0 + c) * DIM + r0 + tx] = tile[c][tx];        // coalesced write
  }
}

// ------------------------------------------------------------------ GEMM ----
// C[4096 x 1024] = alpha * (A[4096 x 1024](bf16) * Bt(bf16, K-major) + bias)
// Block = 256 threads (8 waves); block tile 128x64; wave tile 32x32 (2x2 WMMA).

enum { MODE_F32 = 0, MODE_BF16 = 1, MODE_QK = 2, MODE_VT = 3 };

template <int MODE>
__global__ __launch_bounds__(256)
void gemm4096(const ushort_t* __restrict__ A, const ushort_t* __restrict__ Bt,
              const float* __restrict__ bias, void* __restrict__ Cout,
              float alpha) {
  const int lane  = threadIdx.x & 31;
  const int wave  = threadIdx.x >> 5;
  const int m0    = blockIdx.y * 128 + (wave >> 1) * 32;
  const int n0    = blockIdx.x * 64  + (wave & 1)  * 32;
  const int mlane = lane & 15;
  const int half  = lane >> 4;

  v8f acc[2][2];
#pragma unroll
  for (int i = 0; i < 2; ++i)
#pragma unroll
    for (int j = 0; j < 2; ++j)
#pragma unroll
      for (int r = 0; r < 8; ++r) acc[i][j][r] = 0.0f;

  for (int k0 = 0; k0 < DIM; k0 += 32) {
    Frag a0 = load_afrag(A, DIM, m0,      k0);
    Frag a1 = load_afrag(A, DIM, m0 + 16, k0);
    Frag b0 = load_bfragT(Bt, DIM, n0,      k0);
    Frag b1 = load_bfragT(Bt, DIM, n0 + 16, k0);
    acc[0][0] = wmma_bf16(a0, b0, acc[0][0]);
    acc[0][1] = wmma_bf16(a0, b1, acc[0][1]);
    acc[1][0] = wmma_bf16(a1, b0, acc[1][0]);
    acc[1][1] = wmma_bf16(a1, b1, acc[1][1]);
  }

#pragma unroll
  for (int i = 0; i < 2; ++i)
#pragma unroll
    for (int j = 0; j < 2; ++j) {
      const int n  = n0 + 16 * j + mlane;
      const float bv = bias[n];
#pragma unroll
      for (int r = 0; r < 8; ++r) {
        const int m = m0 + 16 * i + r + 8 * half;   // C layout: lane=N, vgpr=M
        const float val = (acc[i][j][r] + bv) * alpha;
        if (MODE == MODE_F32) {
          ((float*)Cout)[(size_t)m * DIM + n] = val;
        } else if (MODE == MODE_BF16) {
          ((ushort_t*)Cout)[(size_t)m * DIM + n] = f2bf(val);
        } else {
          const int b = m >> 11, s = m & (SEQ - 1);
          const int h = n >> 6,  d = n & 63;
          if (MODE == MODE_QK)   // [B,H,S,Dh]
            ((ushort_t*)Cout)[((((size_t)b * HEADS + h) * SEQ + s) << 6) + d] = f2bf(val);
          else                   // MODE_VT: [B,H,Dh,S]
            ((ushort_t*)Cout)[(((size_t)b * HEADS + h) * DH + d) * SEQ + s] = f2bf(val);
        }
      }
    }
}

// ------------------------------------------------------- flash attention ----
// Q (pre-scaled by 1/sqrt(Dh)), K: [B,H,S,Dh] bf16; Vt: [B,H,Dh,S] bf16;
// O: [B,S,H*Dh] bf16 (row-major).
// Block = 128 threads (4 waves); wave owns 16 q-rows; 32-key blocks.

__global__ __launch_bounds__(128)
void attn_kernel(const ushort_t* __restrict__ Q, const ushort_t* __restrict__ K,
                 const ushort_t* __restrict__ Vt, ushort_t* __restrict__ O) {
  __shared__ ushort_t sP[4][16][32];      // per-wave P staging (C -> A relayout)

  const int lane  = threadIdx.x & 31;
  const int wave  = threadIdx.x >> 5;
  const int mlane = lane & 15;
  const int half  = lane >> 4;
  const int bh = blockIdx.y;              // b*H + h
  const int b  = bh >> 4, h = bh & 15;
  const int q0 = blockIdx.x * 64 + wave * 16;

  const ushort_t* Qh = Q  + (size_t)bh * SEQ * DH;
  const ushort_t* Kh = K  + (size_t)bh * SEQ * DH;
  const ushort_t* Vh = Vt + (size_t)bh * DH * SEQ;

  const Frag aQ0 = load_afrag(Qh, DH, q0, 0);    // d = 0..31
  const Frag aQ1 = load_afrag(Qh, DH, q0, 32);   // d = 32..63

  v8f oacc[4];
  float mrow[8], lrow[8];
#pragma unroll
  for (int j = 0; j < 4; ++j)
#pragma unroll
    for (int r = 0; r < 8; ++r) oacc[j][r] = 0.0f;
#pragma unroll
  for (int r = 0; r < 8; ++r) { mrow[r] = -1e30f; lrow[r] = 0.0f; }

  for (int kt = 0; kt < SEQ; kt += 32) {
    // --- scores: S[16 x 32] as two 16x16 C tiles (Q already scaled) -----
    v8f c0, c1;
#pragma unroll
    for (int r = 0; r < 8; ++r) { c0[r] = 0.0f; c1[r] = 0.0f; }
    {
      Frag bK = load_bfragT(Kh, DH, kt, 0);       c0 = wmma_bf16(aQ0, bK, c0);
      bK      = load_bfragT(Kh, DH, kt, 32);      c0 = wmma_bf16(aQ1, bK, c0);
      bK      = load_bfragT(Kh, DH, kt + 16, 0);  c1 = wmma_bf16(aQ0, bK, c1);
      bK      = load_bfragT(Kh, DH, kt + 16, 32); c1 = wmma_bf16(aQ1, bK, c1);
    }
    // --- online softmax (row stats across 16-lane half-waves) ----------
    float corr[8];
#pragma unroll
    for (int r = 0; r < 8; ++r) {
      const float s0 = c0[r], s1 = c1[r];
      float mx = fmaxf(s0, s1);
#pragma unroll
      for (int msk = 1; msk <= 8; msk <<= 1)
        mx = fmaxf(mx, __shfl_xor(mx, msk, 32));
      const float mnew = fmaxf(mrow[r], mx);
      const float p0 = __expf(s0 - mnew);
      const float p1 = __expf(s1 - mnew);
      float ps = p0 + p1;
#pragma unroll
      for (int msk = 1; msk <= 8; msk <<= 1)
        ps += __shfl_xor(ps, msk, 32);
      corr[r] = __expf(mrow[r] - mnew);
      lrow[r] = lrow[r] * corr[r] + ps;
      mrow[r] = mnew;
      // stage P (C layout -> LDS row-major; wave-private, DScnt-ordered)
      sP[wave][r + 8 * half][mlane]      = f2bf(p0);
      sP[wave][r + 8 * half][16 + mlane] = f2bf(p1);
    }
#pragma unroll
    for (int j = 0; j < 4; ++j)
#pragma unroll
      for (int r = 0; r < 8; ++r) oacc[j][r] *= corr[r];

    // --- reload P as a 16x32 bf16 A-fragment ----------------------------
    Frag aP;
    {
      const ushort_t* row = &sP[wave][mlane][0];
#pragma unroll
      for (int v = 0; v < 8; ++v) {
        int k = (v >> 2) * 16 + half * 8 + (v & 3) * 2;
        aP.u[v] = *(const uint_t*)(row + k);
      }
    }
    // --- O += P @ V  (V supplied K-major via Vt) ------------------------
#pragma unroll
    for (int j = 0; j < 4; ++j) {
      Frag bV = load_bfragT(Vh, SEQ, j * 16, kt);
      oacc[j] = wmma_bf16(aP, bV, oacc[j]);
    }
  }

  // --- normalize and store O in [B,S,H*Dh] ------------------------------
  float linv[8];
#pragma unroll
  for (int r = 0; r < 8; ++r) linv[r] = 1.0f / lrow[r];
#pragma unroll
  for (int j = 0; j < 4; ++j) {
    const int d = j * 16 + mlane;
#pragma unroll
    for (int r = 0; r < 8; ++r) {
      const int s = q0 + r + 8 * half;
      O[(((size_t)(b * SEQ + s)) * HEADS + h) * DH + d] = f2bf(oacc[j][r] * linv[r]);
    }
  }
}

// ------------------------------------------------------------- diff a1-λa2 --

__global__ __launch_bounds__(256)
void diff_kernel(const ushort_t* __restrict__ a1, const ushort_t* __restrict__ a2,
                 const float* __restrict__ lambda_param,
                 const int* __restrict__ layer_idx,
                 ushort_t* __restrict__ d, int n) {
  int i = blockIdx.x * 256 + threadIdx.x;
  if (i >= n) return;
  const float lf  = (float)layer_idx[0];
  const float et  = __expf(-0.3f * fmaxf(lf - 1.0f, 0.0f));
  const float lam = fminf(fmaxf((0.8f - 0.6f * et) * lambda_param[0], 0.1f), 0.9f);
  d[i] = f2bf(bf2f(a1[i]) - lam * bf2f(a2[i]));
}

// ----------------------------------------------------------------- launch ----

extern "C" void kernel_launch(void* const* d_in, const int* in_sizes, int n_in,
                              void* d_out, int out_size, void* d_ws, size_t ws_size,
                              hipStream_t stream) {
  (void)in_sizes; (void)n_in; (void)out_size; (void)ws_size;

  const float* x            = (const float*)d_in[0];
  const int*   layer_idx    = (const int*)  d_in[1];
  const float* lambda_param = (const float*)d_in[2];
  // per-branch: Wq,Wk,Wv,Wo,bq,bk,bv,bo  at 3..10 and 11..18; Wp=19, bp=20
  const float* W[9] = {
    (const float*)d_in[3],  (const float*)d_in[4],  (const float*)d_in[5],
    (const float*)d_in[6],                                          // Wo1
    (const float*)d_in[11], (const float*)d_in[12], (const float*)d_in[13],
    (const float*)d_in[14],                                         // Wo2
    (const float*)d_in[19]                                          // Wp
  };
  const float* bq[2] = { (const float*)d_in[7],  (const float*)d_in[15] };
  const float* bk[2] = { (const float*)d_in[8],  (const float*)d_in[16] };
  const float* bv[2] = { (const float*)d_in[9],  (const float*)d_in[17] };
  const float* bo[2] = { (const float*)d_in[10], (const float*)d_in[18] };
  const float* bp    = (const float*)d_in[20];

  // workspace carve-up (≈90 MB)
  size_t off = 0;
  auto carve = [&](size_t bytes) {
    void* p = (char*)d_ws + off;
    off += (bytes + 255) & ~(size_t)255;
    return p;
  };
  const size_t NTOK = (size_t)MTOT * DIM;          // 4096*1024
  ushort_t* xb = (ushort_t*)carve(NTOK * 2);
  ushort_t* wt[9];
  for (int i = 0; i < 9; ++i) wt[i] = (ushort_t*)carve((size_t)DIM * DIM * 2);
  ushort_t* Qb   = (ushort_t*)carve(NTOK * 2);
  ushort_t* Kb   = (ushort_t*)carve(NTOK * 2);
  ushort_t* Vtb  = (ushort_t*)carve(NTOK * 2);
  ushort_t* O1   = (ushort_t*)carve(NTOK * 2);
  ushort_t* O2   = (ushort_t*)carve(NTOK * 2);
  ushort_t* a1   = (ushort_t*)carve(NTOK * 2);
  ushort_t* a2   = (ushort_t*)carve(NTOK * 2);
  ushort_t* dif  = (ushort_t*)carve(NTOK * 2);

  const dim3 gGemm(DIM / 64, MTOT / 128);          // 16 x 32 blocks, 256 thr
  const dim3 gAttn(SEQ / 64, BATCH * HEADS);       // 32 x 32 blocks, 128 thr

  // 1) precision conversion
  cvt_f32_bf16<<<(int)(NTOK / 256), 256, 0, stream>>>(x, xb, (int)NTOK);
  for (int i = 0; i < 9; ++i)
    tconv1024<<<dim3(32, 32), 256, 0, stream>>>(W[i], wt[i]);

  // 2) both branches: projections + attention (QKV buffers reused)
  ushort_t* Obuf[2] = { O1, O2 };
  for (int br = 0; br < 2; ++br) {
    const int wbase = br * 4;
    gemm4096<MODE_QK><<<gGemm, 256, 0, stream>>>(xb, wt[wbase + 0], bq[br], Qb, SCALE);
    gemm4096<MODE_QK><<<gGemm, 256, 0, stream>>>(xb, wt[wbase + 1], bk[br], Kb, 1.0f);
    gemm4096<MODE_VT><<<gGemm, 256, 0, stream>>>(xb, wt[wbase + 2], bv[br], Vtb, 1.0f);
    attn_kernel<<<gAttn, 128, 0, stream>>>(Qb, Kb, Vtb, Obuf[br]);
  }

  // 3) output projections
  gemm4096<MODE_BF16><<<gGemm, 256, 0, stream>>>(O1, wt[3], bo[0], a1, 1.0f);
  gemm4096<MODE_BF16><<<gGemm, 256, 0, stream>>>(O2, wt[7], bo[1], a2, 1.0f);

  // 4) diff = a1 - lambda * a2
  diff_kernel<<<(int)(NTOK / 256), 256, 0, stream>>>(a1, a2, lambda_param,
                                                     layer_idx, dif, (int)NTOK);

  // 5) out = diff @ Wp + bp  (fp32)
  gemm4096<MODE_F32><<<gGemm, 256, 0, stream>>>(dif, wt[8], bp, (float*)d_out, 1.0f);
}
